// ResilientMLP_28183575396658
// MI455X (gfx1250) — matrix operand order repaired and checked by reading, not compiled, not run on hardware
//
#include <hip/hip_runtime.h>

#define B_DIM 16384
#define D_DIM 1024
#define H_DIM 1024
#define C_DIM 256
#define K2_DIM 960   // H - 2*32
#define F_CO 32

typedef __attribute__((ext_vector_type(16))) __bf16        bf16x16;
typedef __attribute__((ext_vector_type(8)))  float         f32x8;
typedef __attribute__((ext_vector_type(8)))  unsigned int  u32x8;

__device__ __forceinline__ unsigned short f2bf(float f) {
    unsigned int u = __float_as_uint(f);
    unsigned int r = u + 0x7FFFu + ((u >> 16) & 1u);   // round-to-nearest-even
    return (unsigned short)(r >> 16);
}

// order-preserving float->uint key (monotone total order, stable for ties by index)
__device__ __forceinline__ unsigned int f2ord(float f) {
    unsigned int u = __float_as_uint(f);
    return (u & 0x80000000u) ? ~u : (u | 0x80000000u);
}

// ---------------------------------------------------------------- cvt f32->bf16
__global__ __launch_bounds__(256) void cvt_bf16_kernel(const float* __restrict__ src,
                                                       unsigned short* __restrict__ dst,
                                                       int n) {
    int i = blockIdx.x * 256 + threadIdx.x;
    if (i < n) dst[i] = f2bf(src[i]);
}

// ---------------------------------------------------------------- GEMM1 + bias + mask + relu
// h[m,n] = relu((sum_k x[m,k]*W1[n,k] + b1[n]) * mask[m,n])
// grid: (B/32, H/(32*8)), block 256 (8 waves), one 32x32 tile per wave
// (2 A-frags + 2 B-frags -> 4 WMMAs: 1 fragment per WMMA, 16 FLOP/byte from cache).
__global__ __launch_bounds__(256) void gemm1_kernel(const unsigned short* __restrict__ xb,
                                                    const unsigned short* __restrict__ w1b,
                                                    const float* __restrict__ b1,
                                                    const float* __restrict__ mask,
                                                    float* __restrict__ h) {
    const int lane = threadIdx.x & 31;
    const int wave = threadIdx.x >> 5;
    const int m_base = blockIdx.x * 32;
    const int n_base = (blockIdx.y * 8 + wave) * 32;
    const int hf = lane >> 4;     // half-wave
    const int mn = lane & 15;     // M for A-frags / N for B-frags (ISA striping)

    const unsigned short* arow0 = xb  + (size_t)(m_base + mn)       * D_DIM;
    const unsigned short* arow1 = xb  + (size_t)(m_base + 16 + mn)  * D_DIM;
    const unsigned short* brow0 = w1b + (size_t)(n_base + mn)       * D_DIM;
    const unsigned short* brow1 = w1b + (size_t)(n_base + 16 + mn)  * D_DIM;

    f32x8 c00 = {}, c01 = {}, c10 = {}, c11 = {};
    for (int k0 = 0; k0 < D_DIM; k0 += 32) {
        u32x8 au0, au1, bu0, bu1;
#pragma unroll
        for (int r = 0; r < 8; ++r) {
            // A 16x32 16-bit layout: half 0 -> K 0..7,16..23 ; half 1 -> +8
            int ka = (r < 4) ? (2 * r + 8 * hf) : (16 + 2 * (r - 4) + 8 * hf);
            // B 32x16 16-bit layout: same K order as A (half 0 -> K 0..7,16..23 ; half 1 -> +8)
            int kb = (r < 4) ? (2 * r + 8 * hf) : (16 + 2 * (r - 4) + 8 * hf);
            au0[r] = *(const unsigned int*)(arow0 + k0 + ka);
            au1[r] = *(const unsigned int*)(arow1 + k0 + ka);
            bu0[r] = *(const unsigned int*)(brow0 + k0 + kb);
            bu1[r] = *(const unsigned int*)(brow1 + k0 + kb);
        }
        bf16x16 a0 = __builtin_bit_cast(bf16x16, au0);
        bf16x16 a1 = __builtin_bit_cast(bf16x16, au1);
        bf16x16 b0 = __builtin_bit_cast(bf16x16, bu0);
        bf16x16 b1 = __builtin_bit_cast(bf16x16, bu1);
        c00 = __builtin_amdgcn_wmma_f32_16x16x32_bf16(false, a0, false, b0, (short)0, c00, false, false);
        c01 = __builtin_amdgcn_wmma_f32_16x16x32_bf16(false, a0, false, b1, (short)0, c01, false, false);
        c10 = __builtin_amdgcn_wmma_f32_16x16x32_bf16(false, a1, false, b0, (short)0, c10, false, false);
        c11 = __builtin_amdgcn_wmma_f32_16x16x32_bf16(false, a1, false, b1, (short)0, c11, false, false);
    }

    // C/D layout: VGPR r, lanes 0-15 -> M=r ; lanes 16-31 -> M=r+8 ; N=lane%16
#pragma unroll
    for (int mi = 0; mi < 2; ++mi) {
#pragma unroll
        for (int ni = 0; ni < 2; ++ni) {
            f32x8 c = (mi == 0) ? (ni == 0 ? c00 : c01) : (ni == 0 ? c10 : c11);
#pragma unroll
            for (int r = 0; r < 8; ++r) {
                int m = m_base + mi * 16 + r + hf * 8;
                int n = n_base + ni * 16 + mn;
                float v = c[r] + b1[n];
                v *= __builtin_nontemporal_load(&mask[(size_t)m * H_DIM + n]);
                v = v > 0.0f ? v : 0.0f;
                h[(size_t)m * H_DIM + n] = v;
            }
        }
    }
}

// ---------------------------------------------------------------- per-row select + compact
__device__ __forceinline__ unsigned int block_scan_excl(unsigned int myval,
                                                        unsigned int* buf,
                                                        int t, unsigned int* total) {
    buf[t] = myval;
    __syncthreads();
#pragma unroll
    for (int off = 1; off < 256; off <<= 1) {
        unsigned int v = (t >= off) ? buf[t - off] : 0u;
        __syncthreads();
        buf[t] += v;
        __syncthreads();
    }
    unsigned int excl = (t > 0) ? buf[t - 1] : 0u;
    unsigned int tot  = buf[255];
    __syncthreads();   // buf reusable after return
    *total = tot;
    return excl;
}

__global__ __launch_bounds__(256) void select_kernel(const float* __restrict__ h,
                                                     unsigned short* __restrict__ filt) {
    __shared__ float        vals[1024];
    __shared__ unsigned int keys[1024];
    __shared__ unsigned int hist[256];
    __shared__ unsigned int scanbuf[256];
    __shared__ unsigned int s_sel, s_rem;
    __shared__ unsigned int s_cnt[2];

    const int t = threadIdx.x;
    const size_t row = blockIdx.x;
    const float* hrow = h + row * H_DIM;

#pragma unroll
    for (int j = 0; j < 4; ++j) {
        int i = t * 4 + j;
        float v = hrow[i];
        vals[i] = v;
        keys[i] = f2ord(v);
    }
    if (t < 2) s_cnt[t] = 0;
    __syncthreads();

    // exact order statistics via 4-level byte radix select
    unsigned int vsel[2];
    const int ranks[2] = { F_CO - 1, H_DIM - F_CO };   // rank 31 and rank 992 (0-based)
#pragma unroll 1
    for (int s = 0; s < 2; ++s) {
        unsigned int prefix = 0;
        if (t == 0) s_rem = (unsigned int)ranks[s];
        __syncthreads();
        for (int level = 3; level >= 0; --level) {
            const int shift = level * 8;
            hist[t] = 0;
            __syncthreads();
            const unsigned int maskhi = (level == 3) ? 0u : (0xFFFFFFFFu << (shift + 8));
#pragma unroll
            for (int j = 0; j < 4; ++j) {
                unsigned int k = keys[t * 4 + j];
                if ((k & maskhi) == (prefix & maskhi))
                    atomicAdd(&hist[(k >> shift) & 0xFFu], 1u);
            }
            __syncthreads();
            if (t == 0) {
                unsigned int rem = s_rem, c = 0, bsel = 255;
                for (int bin = 0; bin < 256; ++bin) {
                    unsigned int nc = c + hist[bin];
                    if (rem < nc) { bsel = (unsigned int)bin; break; }
                    c = nc;
                }
                s_sel = bsel;
                s_rem = rem - c;
            }
            __syncthreads();
            prefix |= (s_sel << shift);
            __syncthreads();
        }
        vsel[s] = prefix;
    }
    const unsigned int v_lo = vsel[0], v_hi = vsel[1];

    // counts + packed eq-prefix scan (eq_lo in [15:0], eq_hi in [31:16])
    unsigned int my_lt = 0, my_gt = 0;
    unsigned int locpref[4], run = 0;
#pragma unroll
    for (int j = 0; j < 4; ++j) {
        unsigned int k = keys[t * 4 + j];
        locpref[j] = run;
        run += (k == v_lo ? 1u : 0u) | (k == v_hi ? 0x10000u : 0u);
        my_lt += (k < v_lo) ? 1u : 0u;
        my_gt += (k > v_hi) ? 1u : 0u;
    }
    atomicAdd(&s_cnt[0], my_lt);
    atomicAdd(&s_cnt[1], my_gt);
    unsigned int tot_pack;
    unsigned int excl_pack = block_scan_excl(run, scanbuf, t, &tot_pack);
    __syncthreads();
    const unsigned int c_lt = s_cnt[0], c_gt = s_cnt[1];
    const unsigned int c_eq_hi  = tot_pack >> 16;
    const unsigned int n_drop_lo = F_CO - c_lt;   // ties dropped at low end (smallest indices)
    const unsigned int n_drop_hi = F_CO - c_gt;   // ties dropped at high end (largest indices)

    // keep flags (stable, matches argsort tie order) + compaction positions
    int keep[4]; unsigned int kpref[4], krun = 0;
#pragma unroll
    for (int j = 0; j < 4; ++j) {
        unsigned int k = keys[t * 4 + j];
        unsigned int p = excl_pack + locpref[j];
        unsigned int eqlo_before = p & 0xFFFFu;
        unsigned int eqhi_incl   = (p >> 16) + (k == v_hi ? 1u : 0u);
        unsigned int eqhi_after  = c_eq_hi - eqhi_incl;
        int kp;
        if (k < v_lo || k > v_hi)                          kp = 0;
        else if (k == v_lo && eqlo_before < n_drop_lo)     kp = 0;
        else if (k == v_hi && eqhi_after  < n_drop_hi)     kp = 0;
        else                                               kp = 1;
        keep[j] = kp;
        kpref[j] = krun;
        krun += (unsigned int)kp;
    }
    unsigned int tot_keep;
    unsigned int excl_keep = block_scan_excl(krun, scanbuf, t, &tot_keep);
    (void)tot_keep; // == 960
#pragma unroll
    for (int j = 0; j < 4; ++j) {
        if (keep[j]) {
            unsigned int pos = excl_keep + kpref[j];
            filt[row * K2_DIM + pos] = f2bf(vals[t * 4 + j]);
        }
    }
}

// ---------------------------------------------------------------- GEMM2 + bias + softmax
// grid: B/16 blocks; block 256 = 8 waves; wave w owns columns [32w, 32w+32).
__global__ __launch_bounds__(256) void gemm2_softmax_kernel(const unsigned short* __restrict__ filt,
                                                            const unsigned short* __restrict__ w2b,
                                                            const float* __restrict__ b2,
                                                            float* __restrict__ out) {
    __shared__ float lg[16][C_DIM];

    const int lane = threadIdx.x & 31;
    const int wave = threadIdx.x >> 5;
    const int m_base = blockIdx.x * 16;
    const int hf = lane >> 4;
    const int mn = lane & 15;

    const unsigned short* arow  = filt + (size_t)(m_base + mn) * K2_DIM;
    const unsigned short* brow0 = w2b  + (size_t)(wave * 32 + mn)      * K2_DIM;
    const unsigned short* brow1 = w2b  + (size_t)(wave * 32 + 16 + mn) * K2_DIM;

    f32x8 c0 = {}, c1 = {};
    for (int k0 = 0; k0 < K2_DIM; k0 += 32) {
        u32x8 au, bu0, bu1;
#pragma unroll
        for (int r = 0; r < 8; ++r) {
            int ka = (r < 4) ? (2 * r + 8 * hf) : (16 + 2 * (r - 4) + 8 * hf);
            int kb = (r < 4) ? (2 * r + 8 * hf) : (16 + 2 * (r - 4) + 8 * hf);
            au[r]  = *(const unsigned int*)(arow  + k0 + ka);
            bu0[r] = *(const unsigned int*)(brow0 + k0 + kb);
            bu1[r] = *(const unsigned int*)(brow1 + k0 + kb);
        }
        bf16x16 a = __builtin_bit_cast(bf16x16, au);
        c0 = __builtin_amdgcn_wmma_f32_16x16x32_bf16(false, a, false,
                 __builtin_bit_cast(bf16x16, bu0), (short)0, c0, false, false);
        c1 = __builtin_amdgcn_wmma_f32_16x16x32_bf16(false, a, false,
                 __builtin_bit_cast(bf16x16, bu1), (short)0, c1, false, false);
    }
#pragma unroll
    for (int r = 0; r < 8; ++r) {
        int m = r + hf * 8;
        int n0 = wave * 32 + mn;
        int n1 = n0 + 16;
        lg[m][n0] = c0[r] + b2[n0];
        lg[m][n1] = c1[r] + b2[n1];
    }
    __syncthreads();

    // fused softmax: each wave handles 2 rows of 256 logits (8 per lane)
#pragma unroll 1
    for (int rr = 0; rr < 2; ++rr) {
        int m = wave * 2 + rr;
        float v[8];
        float mx = -3.402823466e38f;
#pragma unroll
        for (int j = 0; j < 8; ++j) { v[j] = lg[m][lane + 32 * j]; mx = fmaxf(mx, v[j]); }
#pragma unroll
        for (int off = 16; off >= 1; off >>= 1) mx = fmaxf(mx, __shfl_xor(mx, off, 32));
        float sum = 0.0f;
#pragma unroll
        for (int j = 0; j < 8; ++j) { v[j] = __expf(v[j] - mx); sum += v[j]; }
#pragma unroll
        for (int off = 16; off >= 1; off >>= 1) sum += __shfl_xor(sum, off, 32);
        float inv = 1.0f / sum;
#pragma unroll
        for (int j = 0; j < 8; ++j)
            out[(size_t)(m_base + m) * C_DIM + lane + 32 * j] = v[j] * inv;
    }
}

// ---------------------------------------------------------------- launch
extern "C" void kernel_launch(void* const* d_in, const int* in_sizes, int n_in,
                              void* d_out, int out_size, void* d_ws, size_t ws_size,
                              hipStream_t stream) {
    const float* x    = (const float*)d_in[0];
    const float* W1   = (const float*)d_in[1];
    const float* b1   = (const float*)d_in[2];
    const float* W2   = (const float*)d_in[3];
    const float* b2   = (const float*)d_in[4];
    const float* mask = (const float*)d_in[5];
    float* out = (float*)d_out;

    char* ws = (char*)d_ws;
    size_t off = 0;
    unsigned short* xb  = (unsigned short*)(ws + off); off += (size_t)B_DIM * D_DIM * 2;   // 32 MB
    unsigned short* w1b = (unsigned short*)(ws + off); off += (size_t)H_DIM * D_DIM * 2;   // 2 MB
    unsigned short* w2b = (unsigned short*)(ws + off); off += (size_t)C_DIM * K2_DIM * 2;  // 480 KB
    float*          hbf = (float*)(ws + off);          off += (size_t)B_DIM * H_DIM * 4;   // 64 MB
    unsigned short* flt = (unsigned short*)(ws + off); off += (size_t)B_DIM * K2_DIM * 2;  // 30 MB

    // 1) convert operands to bf16
    {
        int n1 = B_DIM * D_DIM;
        cvt_bf16_kernel<<<(n1 + 255) / 256, 256, 0, stream>>>(x, xb, n1);
        int n2 = H_DIM * D_DIM;
        cvt_bf16_kernel<<<(n2 + 255) / 256, 256, 0, stream>>>(W1, w1b, n2);
        int n3 = C_DIM * K2_DIM;
        cvt_bf16_kernel<<<(n3 + 255) / 256, 256, 0, stream>>>(W2, w2b, n3);
    }
    // 2) fc1 + dropout-mask + relu (32x32 tile per wave, 4 WMMAs per fragment-quad)
    gemm1_kernel<<<dim3(B_DIM / 32, H_DIM / (32 * 8)), 256, 0, stream>>>(xb, w1b, b1, mask, hbf);
    // 3) resilient select + compact
    select_kernel<<<B_DIM, 256, 0, stream>>>(hbf, flt);
    // 4) fc2 + softmax
    gemm2_softmax_kernel<<<B_DIM / 16, 256, 0, stream>>>(flt, w2b, b2, out);
}
